// Decoder_79242146611500
// MI455X (gfx1250) — matrix operand order repaired and checked by reading, not compiled, not run on hardware
//
#include <hip/hip_runtime.h>
#include <stdint.h>

// ---------------- CDNA5 WMMA types ----------------
typedef __attribute__((ext_vector_type(16))) __bf16 v16bf;
typedef __attribute__((ext_vector_type(8)))  float  v8f;

union FragB { v16bf v; uint4 q[2]; };

// ---------------- problem constants ----------------
constexpr int Bb = 2048, Tt = 250, Dd = 128, Hh = 100;
constexpr int ROWS = 16;              // batch rows per workgroup (one M tile)
constexpr int GN   = 400;             // 4*H gate width
constexpr int NT   = 25;              // GN/16 N-tiles for the gate GEMM
// LDS strides (elements), padded to avoid 64-bank conflicts
constexpr int WT_STRIDE  = 264;       // stacked [Wi;Wh;0] transposed: [400][256]+pad bf16
constexpr int WDT_STRIDE = 136;       // Wd transposed: [112][128]+pad bf16
constexpr int XH_STRIDE  = 264;       // [x(128)|h(100)|0] per row, bf16
constexpr int Z_STRIDE   = 404;       // z buffer [16][400]+pad f32
// LDS byte offsets
constexpr int OFF_WT  = 0;                                  // 211200 B
constexpr int OFF_WDT = OFF_WT  + 400 * WT_STRIDE * 2;      //  30464 B
constexpr int OFF_XH  = OFF_WDT + 112 * WDT_STRIDE * 2;     //   8448 B
constexpr int OFF_Z   = OFF_XH  + ROWS * XH_STRIDE * 2;     //  25856 B
constexpr int OFF_C   = OFF_Z   + ROWS * Z_STRIDE * 4;      //   6400 B
constexpr int OFF_BH  = OFF_C   + ROWS * Hh * 4;            //   1600 B
constexpr int OFF_BD  = OFF_BH  + GN * 4;                   //    448 B
constexpr int SMEM_BYTES = OFF_BD + 112 * 4;                // 284416 B < 320KB WGP LDS

// ---------------- helpers ----------------
__device__ __forceinline__ unsigned short f2bf(float f) {
  uint32_t u = __builtin_bit_cast(uint32_t, f);
  uint32_t r = u + 0x7FFFu + ((u >> 16) & 1u);   // round-to-nearest-even
  return (unsigned short)(r >> 16);
}
__device__ __forceinline__ float fsig(float x) {
  return __builtin_amdgcn_rcpf(1.0f + __builtin_amdgcn_exp2f(-1.4426950408889634f * x));
}
__device__ __forceinline__ float ftanh(float x) {
  return 2.0f * __builtin_amdgcn_rcpf(1.0f + __builtin_amdgcn_exp2f(-2.8853900817779268f * x)) - 1.0f;
}

// A fragment (16x32 bf16, M-major rows in LDS).
// ISA layout: lanes 0-15 M=0-15 hold K {0..7,16..23}; lanes 16-31 hold K {8..15,24..31}.
__device__ __forceinline__ v16bf load_a(const char* xh, int lane, int kbase) {
  int row = lane & 15, hi = (lane >> 4) & 1;
  const char* p = xh + row * (XH_STRIDE * 2) + (kbase + 8 * hi) * 2;
  FragB f;
  f.q[0] = *(const uint4*)p;          // K = kbase+8*hi .. +7
  f.q[1] = *(const uint4*)(p + 32);   // K = kbase+16+8*hi .. +7
  return f.v;
}
// B fragment (32x16 bf16) from N-major (transposed) weight storage.
// ISA layout: lane = N (mod 16); lanes 0-15 K {0..7,16..23}; lanes 16-31 K {8..15,24..31}.
__device__ __forceinline__ v16bf load_b(const char* wt, int lane, int ncol0,
                                        int kbase, int strideBytes) {
  int n = lane & 15, hi = (lane >> 4) & 1;
  const char* p = wt + (ncol0 + n) * strideBytes + (kbase + 8 * hi) * 2;
  FragB f;
  f.q[0] = *(const uint4*)p;          // K = kbase+8*hi .. +7
  f.q[1] = *(const uint4*)(p + 32);   // K = kbase+16+8*hi .. +7
  return f.v;
}

#define WMMA_BF16(A, Bm, C) \
  __builtin_amdgcn_wmma_f32_16x16x32_bf16(false, (A), false, (Bm), (short)0, (C), false, false)

// ---------------- fused persistent LSTM kernel ----------------
extern "C" __global__ __launch_bounds__(512, 1)
void lstm_fused(const float* __restrict__ x,  const float* __restrict__ Wi,
                const float* __restrict__ Wh, const float* __restrict__ bh,
                const float* __restrict__ Wd, const float* __restrict__ bd,
                float* __restrict__ out) {
  extern __shared__ char smem[];
  char*  sWT  = smem + OFF_WT;
  char*  sWDT = smem + OFF_WDT;
  char*  sXH  = smem + OFF_XH;
  float* sZ   = (float*)(smem + OFF_Z);
  float* sC   = (float*)(smem + OFF_C);
  float* sBH  = (float*)(smem + OFF_BH);
  float* sBD  = (float*)(smem + OFF_BD);

  const int tid   = threadIdx.x;
  const int lane  = tid & 31;
  const int wave  = tid >> 5;          // 0..15
  const int bbase = blockIdx.x * ROWS;

  // ---- one-time LDS init: stacked transposed weights WT[n][k] ----
  // k<128 -> Wi[k][n]; 128<=k<228 -> Wh[k-128][n]; 228..255 -> 0 (h padding)
  for (int idx = tid; idx < 400 * 256; idx += 512) {
    int n = idx >> 8, k = idx & 255;
    float v = 0.f;
    if (k < 128)      v = Wi[k * 400 + n];
    else if (k < 228) v = Wh[(k - 128) * 400 + n];
    *(unsigned short*)(sWT + n * (WT_STRIDE * 2) + k * 2) = f2bf(v);
  }
  // Wd transposed, zero-padded to [112][128]
  for (int idx = tid; idx < 112 * 128; idx += 512) {
    int n = idx >> 7, k = idx & 127;
    float v = (n < Hh && k < Hh) ? Wd[k * Hh + n] : 0.f;
    *(unsigned short*)(sWDT + n * (WDT_STRIDE * 2) + k * 2) = f2bf(v);
  }
  for (int idx = tid; idx < GN; idx += 512) sBH[idx] = bh[idx];
  if (tid < 112) sBD[tid] = (tid < Hh) ? bd[tid] : 0.f;
  for (int idx = tid; idx < ROWS * Hh; idx += 512) sC[idx] = 0.f;    // c0 = 0
  for (int idx = tid; idx < ROWS * 128; idx += 512) {                // h0 = 0 (+pad)
    int row = idx >> 7, k = 128 + (idx & 127);
    *(unsigned short*)(sXH + row * (XH_STRIDE * 2) + k * 2) = 0;
  }
  __syncthreads();

  // ---- hoist invariant B fragments into VGPRs (live across the whole scan) ----
  const int  tile0   = wave;                 // N-tiles of z: wave and wave+16
  const int  tile1i  = wave + 16;
  const bool has1    = (tile1i < NT);
  const int  tile1   = has1 ? tile1i : tile0;
  v16bf wf0[8], wf1[8];
  #pragma unroll
  for (int kb = 0; kb < 8; ++kb) wf0[kb] = load_b(sWT, lane, tile0 * 16, kb * 32, WT_STRIDE * 2);
  #pragma unroll
  for (int kb = 0; kb < 8; ++kb) wf1[kb] = load_b(sWT, lane, tile1 * 16, kb * 32, WT_STRIDE * 2);
  const bool hasProj = (wave < 7);           // 7 tiles cover 112 >= H output cols
  const int  ptile   = hasProj ? wave : 0;
  v16bf wdf[4];
  #pragma unroll
  for (int kb = 0; kb < 4; ++kb) wdf[kb] = load_b(sWDT, lane, ptile * 16, kb * 32, WDT_STRIDE * 2);

  // per-thread x-tile loader coordinates: one float4 per thread per step
  const int xrow = tid >> 5;                  // 0..15
  const int xcol = (tid & 31) * 4;            // 0..124
  const float* xptr = x + ((size_t)(bbase + xrow) * Tt) * Dd + xcol;
  char* xhw = sXH + xrow * (XH_STRIDE * 2) + xcol * 2;

  const int hi = (lane >> 4) & 1;
  const int nl = lane & 15;

  for (int t = 0; t < Tt; ++t) {
    // (a) stream next x tile into the A-buffer (cols 0..127), fp32 -> bf16
    float4 xv = *(const float4*)(xptr + (size_t)t * Dd);
    if (t + 1 < Tt) __builtin_prefetch(xptr + (size_t)(t + 1) * Dd, 0, 1);
    uint32_t p0 = (uint32_t)f2bf(xv.x) | ((uint32_t)f2bf(xv.y) << 16);
    uint32_t p1 = (uint32_t)f2bf(xv.z) | ((uint32_t)f2bf(xv.w) << 16);
    uint2 pk; pk.x = p0; pk.y = p1;
    *(uint2*)xhw = pk;
    __syncthreads();

    // (b) fused gate GEMM: z = [x_t | h] @ [Wi ; Wh]  (K = 256, bf16 WMMA, fp32 acc)
    {
      v8f acc0 = {0.f, 0.f, 0.f, 0.f, 0.f, 0.f, 0.f, 0.f};
      #pragma unroll
      for (int kb = 0; kb < 8; ++kb) {
        v16bf a = load_a(sXH, lane, kb * 32);
        acc0 = WMMA_BF16(a, wf0[kb], acc0);
      }
      int nc = tile0 * 16 + nl;
      #pragma unroll
      for (int r = 0; r < 8; ++r) sZ[(r + 8 * hi) * Z_STRIDE + nc] = acc0[r];
    }
    if (has1) {
      v8f acc1 = {0.f, 0.f, 0.f, 0.f, 0.f, 0.f, 0.f, 0.f};
      #pragma unroll
      for (int kb = 0; kb < 8; ++kb) {
        v16bf a = load_a(sXH, lane, kb * 32);
        acc1 = WMMA_BF16(a, wf1[kb], acc1);
      }
      int nc = tile1 * 16 + nl;
      #pragma unroll
      for (int r = 0; r < 8; ++r) sZ[(r + 8 * hi) * Z_STRIDE + nc] = acc1[r];
    }
    __syncthreads();

    // (c) gates + state update (i,f,g,o order); h written back as bf16 into A-buffer
    for (int idx = tid; idx < ROWS * Hh; idx += 512) {
      int row = idx / Hh, hc = idx % Hh;
      const float* zr = sZ + row * Z_STRIDE;
      float ig = fsig (zr[hc]        + sBH[hc]);
      float fg = fsig (zr[100 + hc]  + sBH[100 + hc]);
      float gg = ftanh(zr[200 + hc]  + sBH[200 + hc]);
      float og = fsig (zr[300 + hc]  + sBH[300 + hc]);
      float c  = fg * sC[idx] + ig * gg;
      sC[idx]  = c;
      float h  = og * ftanh(c);
      *(unsigned short*)(sXH + row * (XH_STRIDE * 2) + (128 + hc) * 2) = f2bf(h);
    }
    __syncthreads();

    // (d) fused output projection + bias + ReLU, streamed straight to HBM
    if (hasProj) {
      v8f acc = {0.f, 0.f, 0.f, 0.f, 0.f, 0.f, 0.f, 0.f};
      #pragma unroll
      for (int kb = 0; kb < 4; ++kb) {
        v16bf a = load_a(sXH, lane, 128 + kb * 32);   // h part of the A-buffer
        acc = WMMA_BF16(a, wdf[kb], acc);
      }
      int nc = ptile * 16 + nl;
      if (nc < Hh) {
        float bb = sBD[nc];
        #pragma unroll
        for (int r = 0; r < 8; ++r) {
          int m = r + 8 * hi;
          float v = acc[r] + bb;
          v = v > 0.f ? v : 0.f;
          out[((size_t)(bbase + m) * Tt + t) * Hh + nc] = v;
        }
      }
    }
    // no barrier needed: (d)/next (a) touch disjoint LDS regions; next barrier
    // is inside the next iteration before any cross-region reads.
  }
}

// ---------------- host launcher ----------------
extern "C" void kernel_launch(void* const* d_in, const int* in_sizes, int n_in,
                              void* d_out, int out_size, void* d_ws, size_t ws_size,
                              hipStream_t stream) {
  (void)in_sizes; (void)n_in; (void)out_size; (void)d_ws; (void)ws_size;
  const float* x  = (const float*)d_in[0];
  const float* Wi = (const float*)d_in[1];
  const float* Wh = (const float*)d_in[2];
  const float* bh = (const float*)d_in[3];
  const float* Wd = (const float*)d_in[4];
  const float* bd = (const float*)d_in[5];
  float* out = (float*)d_out;

  (void)hipFuncSetAttribute((const void*)lstm_fused,
                            hipFuncAttributeMaxDynamicSharedMemorySize, SMEM_BYTES);
  lstm_fused<<<Bb / ROWS, 512, SMEM_BYTES, stream>>>(x, Wi, Wh, bh, Wd, bd, out);
}
